// WindowAttention5D_26130581028952
// MI455X (gfx1250) — compile-verified, matrix operand order checked by reading, not run on hardware
//
#include <hip/hip_runtime.h>

typedef __attribute__((ext_vector_type(16))) _Float16 v16h;
typedef __attribute__((ext_vector_type(8)))  _Float16 v8h;
typedef __attribute__((ext_vector_type(8)))  float    v8f;
typedef __attribute__((ext_vector_type(4)))  float    v4f;

#define DIMC   192
#define HEADS  6
#define HD     32
#define NTOK   64
#define QSCALE 0.17677669529663687f   // 32^-0.5

// LDS strides (in halfs / floats), all multiples of 8 for 16B-aligned b128 LDS ops
#define XS  200   // x / attn-out buffer stride
#define QS  592   // qkv buffer stride
#define VTS 72    // transposed-V stride
#define PS  72    // probs stride
#define MS  68    // mask (f32) stride

// LDS byte offsets
#define OFF_X    0
#define OFF_QKV  25600      // 64*XS*2
#define OFF_VT   101376     // + 64*QS*2
#define OFF_P    129024     // + 192*VTS*2
#define OFF_M    184320     // + 6*64*PS*2
#define LDS_TOT  201728     // + 64*MS*4

__device__ __forceinline__ v8f wmma_f16(v16h a, v16h b, v8f c) {
    return __builtin_amdgcn_wmma_f32_16x16x32_f16(false, a, false, b, (short)0, c, false, false);
}

// Load a 16x32 f16 fragment (A-layout; also serves as B when source rows are B's columns)
// ISA layout: lanes 0-15 row r hold K = kbase+{0..7, 16..23}; lanes 16-31 K = kbase+{8..15, 24..31}
__device__ __forceinline__ v16h lds_frag(const _Float16* base, int row, int rs, int kbase) {
    const int lane = threadIdx.x & 31;
    const int g = lane >> 4, r = lane & 15;
    const _Float16* p = base + (size_t)(row + r) * rs + kbase + g * 8;
    v8h lo = *(const v8h*)p;          // ds_load_b128
    v8h hi = *(const v8h*)(p + 16);   // ds_load_b128
    v16h o;
#pragma unroll
    for (int i = 0; i < 8; i++) { o[i] = lo[i]; o[i + 8] = hi[i]; }
    return o;
}

// B fragment straight from global f32 weights (row d of W == column n of B, K contiguous)
__device__ __forceinline__ v16h ldg_fragB(const float* base, int row, int rs, int kbase) {
    const int lane = threadIdx.x & 31;
    const int g = lane >> 4, n = lane & 15;
    const float* p = base + (size_t)(row + n) * rs + kbase + g * 8;
    v4f a0 = *(const v4f*)p;
    v4f a1 = *(const v4f*)(p + 4);
    v4f a2 = *(const v4f*)(p + 16);
    v4f a3 = *(const v4f*)(p + 20);
    v16h o;
#pragma unroll
    for (int i = 0; i < 4; i++) {
        o[i]      = (_Float16)a0[i];
        o[4 + i]  = (_Float16)a1[i];
        o[8 + i]  = (_Float16)a2[i];
        o[12 + i] = (_Float16)a3[i];
    }
    return o;
}

extern "C" __global__ __launch_bounds__(256)
void win_attn_fused(const float* __restrict__ x, const float* __restrict__ mask,
                    const float* __restrict__ wqkv, const float* __restrict__ wproj,
                    const float* __restrict__ bproj, float* __restrict__ y) {
    extern __shared__ char smem[];
    _Float16* xh   = (_Float16*)(smem + OFF_X);    // 64  x XS  (later aliased as attn-out)
    _Float16* qkvh = (_Float16*)(smem + OFF_QKV);  // 64  x QS  (cols 0..575 = q|k|v)
    _Float16* vT   = (_Float16*)(smem + OFF_VT);   // 192 x VTS (V transposed: [d][token])
    _Float16* pb   = (_Float16*)(smem + OFF_P);    // 6 x 64 x PS (per-head probs)
    float*    mlds = (float*)   (smem + OFF_M);    // 64  x MS

    const int bw   = blockIdx.x;
    const int tid  = threadIdx.x;
    const int wid  = tid >> 5;
    const int lane = tid & 31;
    const int ln   = lane & 15;
    const int m0   = (lane >> 4) * 8;   // C/D layout: VGPR r holds row m0+r, col ln

    const float* xg = x    + (size_t)bw * (NTOK * DIMC);
    const float* mg = mask + (size_t)bw * (NTOK * NTOK);

    // ---------------- Stage 0: stage x (f16) and mask (f32) into LDS ----------------
    for (int i = tid; i < NTOK * DIMC / 4; i += 256) {
        v4f v = ((const v4f*)xg)[i];
        int rr = (i * 4) / DIMC, cc = (i * 4) % DIMC;
        _Float16* d = xh + rr * XS + cc;
        d[0] = (_Float16)v[0]; d[1] = (_Float16)v[1];
        d[2] = (_Float16)v[2]; d[3] = (_Float16)v[3];
    }
    for (int i = tid; i < NTOK * NTOK / 4; i += 256) {
        v4f v = ((const v4f*)mg)[i];
        int rr = (i * 4) / NTOK, cc = (i * 4) % NTOK;
        float* d = mlds + rr * MS + cc;
        d[0] = v[0]; d[1] = v[1]; d[2] = v[2]; d[3] = v[3];
    }
    __syncthreads();

    // ---------------- Stage 1: qkv = x @ Wqkv^T  (64x192 · 192x576) ----------------
    for (int nt = wid; nt < 36; nt += 8) {
        const int d0 = nt * 16;
        v16h bf[6];
#pragma unroll
        for (int kk = 0; kk < 6; kk++) bf[kk] = ldg_fragB(wqkv, d0, DIMC, kk * 32);
        const float scl = (d0 < DIMC) ? QSCALE : 1.0f;   // pre-scale q
#pragma unroll
        for (int mt = 0; mt < 4; mt++) {
            v8f acc = {};
#pragma unroll
            for (int kk = 0; kk < 6; kk++) {
                v16h a = lds_frag(xh, mt * 16, XS, kk * 32);
                acc = wmma_f16(a, bf[kk], acc);
            }
            _Float16* dst = qkvh + (size_t)(mt * 16 + m0) * QS + d0 + ln;
#pragma unroll
            for (int r = 0; r < 8; r++) dst[(size_t)r * QS] = (_Float16)(acc[r] * scl);
            if (d0 >= 2 * DIMC) {  // V region: also store transposed for the PV GEMM
#pragma unroll
                for (int r = 0; r < 8; r++)
                    vT[(size_t)(d0 - 2 * DIMC + ln) * VTS + mt * 16 + m0 + r] = (_Float16)acc[r];
            }
        }
    }
    __syncthreads();

    // ---------------- Stage 2: scores = qK^T + mask ; softmax -> probs (f16) ----------------
    for (int c = wid; c < 24; c += 8) {
        const int h = c >> 2, mt = c & 3;
        v16h aq = lds_frag(qkvh + h * HD, mt * 16, QS, 0);          // Q tile 16x32
        v8f s[4];
#pragma unroll
        for (int nt = 0; nt < 4; nt++) {
            v16h bk = lds_frag(qkvh + DIMC + h * HD, nt * 16, QS, 0); // K rows = B cols
            v8f z = {};
            s[nt] = wmma_f16(aq, bk, z);
        }
        const float* mrow = mlds + (size_t)(mt * 16 + m0) * MS + ln;
#pragma unroll
        for (int r = 0; r < 8; r++) {
            float v0 = s[0][r] + mrow[r * MS +  0];
            float v1 = s[1][r] + mrow[r * MS + 16];
            float v2 = s[2][r] + mrow[r * MS + 32];
            float v3 = s[3][r] + mrow[r * MS + 48];
            float mx = fmaxf(fmaxf(v0, v1), fmaxf(v2, v3));
            mx = fmaxf(mx, __shfl_xor(mx, 1));
            mx = fmaxf(mx, __shfl_xor(mx, 2));
            mx = fmaxf(mx, __shfl_xor(mx, 4));
            mx = fmaxf(mx, __shfl_xor(mx, 8));   // row lives in a 16-lane group
            v0 = __expf(v0 - mx); v1 = __expf(v1 - mx);
            v2 = __expf(v2 - mx); v3 = __expf(v3 - mx);
            float ss = v0 + v1 + v2 + v3;
            ss += __shfl_xor(ss, 1); ss += __shfl_xor(ss, 2);
            ss += __shfl_xor(ss, 4); ss += __shfl_xor(ss, 8);
            float inv = 1.0f / ss;
            s[0][r] = v0 * inv; s[1][r] = v1 * inv;
            s[2][r] = v2 * inv; s[3][r] = v3 * inv;
        }
        _Float16* ph = pb + (size_t)h * 64 * PS;
#pragma unroll
        for (int nt = 0; nt < 4; nt++) {
            _Float16* dst = ph + (size_t)(mt * 16 + m0) * PS + nt * 16 + ln;
#pragma unroll
            for (int r = 0; r < 8; r++) dst[(size_t)r * PS] = (_Float16)s[nt][r];
        }
    }
    __syncthreads();

    // ---------------- Stage 3: out = probs @ V  (64x64 · 64x32 per head) ----------------
    _Float16* outh = xh;  // alias: x no longer needed
    for (int c = wid; c < 24; c += 8) {
        const int h = c >> 2, mt = c & 3;
        const _Float16* ph = pb + (size_t)h * 64 * PS;
#pragma unroll
        for (int nt2 = 0; nt2 < 2; nt2++) {
            v8f acc = {};
#pragma unroll
            for (int kk = 0; kk < 2; kk++) {
                v16h a = lds_frag(ph, mt * 16, PS, kk * 32);
                v16h b = lds_frag(vT, h * HD + nt2 * 16, VTS, kk * 32); // vT rows = B cols
                acc = wmma_f16(a, b, acc);
            }
            _Float16* dst = outh + (size_t)(mt * 16 + m0) * XS + h * HD + nt2 * 16 + ln;
#pragma unroll
            for (int r = 0; r < 8; r++) dst[(size_t)r * XS] = (_Float16)acc[r];
        }
    }
    __syncthreads();

    // ---------------- Stage 4: y = out @ Wproj^T + b ----------------
    float* yg = y + (size_t)bw * (NTOK * DIMC);
    for (int t = wid; t < 48; t += 8) {
        const int mt = t & 3, nt = t >> 2;
        v16h bf[6];
#pragma unroll
        for (int kk = 0; kk < 6; kk++) bf[kk] = ldg_fragB(wproj, nt * 16, DIMC, kk * 32);
        v8f acc = {};
#pragma unroll
        for (int kk = 0; kk < 6; kk++) {
            v16h a = lds_frag(outh, mt * 16, XS, kk * 32);
            acc = wmma_f16(a, bf[kk], acc);
        }
        const float bias = bproj[nt * 16 + ln];
        float* dst = yg + (size_t)(mt * 16 + m0) * DIMC + nt * 16 + ln;
#pragma unroll
        for (int r = 0; r < 8; r++) dst[(size_t)r * DIMC] = acc[r] + bias;
    }
}

extern "C" void kernel_launch(void* const* d_in, const int* in_sizes, int n_in,
                              void* d_out, int out_size, void* d_ws, size_t ws_size,
                              hipStream_t stream) {
    (void)in_sizes; (void)n_in; (void)out_size; (void)d_ws; (void)ws_size;
    const float* x     = (const float*)d_in[0];
    const float* mask  = (const float*)d_in[1];
    const float* wqkv  = (const float*)d_in[2];
    const float* wproj = (const float*)d_in[3];
    const float* bproj = (const float*)d_in[4];
    float* y = (float*)d_out;

    hipFuncSetAttribute((const void*)win_attn_fused,
                        hipFuncAttributeMaxDynamicSharedMemorySize, LDS_TOT);
    win_attn_fused<<<4096, 256, LDS_TOT, stream>>>(x, mask, wqkv, wproj, bproj, y);
}
